// NeuralHMM_16003048145386
// MI455X (gfx1250) — matrix operand order repaired and checked by reading, not compiled, not run on hardware
//
#include <hip/hip_runtime.h>
#include <math.h>

#define NSTATES 32
#define INSZ    128
#define HID     256
#define BB      32
#define TT      8192

typedef _Float16 v16h __attribute__((ext_vector_type(16)));
typedef _Float16 v8h  __attribute__((ext_vector_type(8)));
typedef float    v8f  __attribute__((ext_vector_type(8)));
typedef float    v4f  __attribute__((ext_vector_type(4)));

__device__ __forceinline__ v16h joinh(v8h lo, v8h hi) {
    union { v16h v; v8h h[2]; } u;
    u.h[0] = lo; u.h[1] = hi;
    return u.v;
}

__device__ __forceinline__ float gelu_exact(float x) {
    return 0.5f * x * (1.0f + erff(x * 0.70710678118654752f));
}

// ---------------------------------------------------------------------------
// Swizzle an f32 row-major KxN weight matrix into 16x16x32 f16 B-fragment
// order: tile (kt,nt), lane = (n%16) + 16*((k%16)/8), half = k%8 + 8*((k%32)/16).
// ---------------------------------------------------------------------------
__global__ void swizzleW(const float* __restrict__ W, _Float16* __restrict__ out,
                         int K, int N) {
    int idx = blockIdx.x * blockDim.x + threadIdx.x;
    if (idx >= K * N) return;
    int k = idx / N, n = idx % N;
    int kt = k >> 5, nt = n >> 4;
    int lane = (n & 15) | (((k >> 3) & 1) << 4);
    int h = (k & 7) | (((k >> 4) & 1) << 3);
    int NT = N >> 4;
    out[(((kt * NT + nt) * 32 + lane) << 4) + h] = (_Float16)W[idx];
}

// ---------------------------------------------------------------------------
// Build trans (softmax rows), log_trans, log_init. One wave, lane = column.
// ---------------------------------------------------------------------------
__global__ __launch_bounds__(32) void prep_hmm(const float* __restrict__ il,
                                               const float* __restrict__ tl,
                                               float* __restrict__ tr,
                                               float* __restrict__ logT,
                                               float* __restrict__ logI) {
    int j = threadIdx.x;
    {
        float v = il[j];
        float mx = v;
        for (int m = 1; m < 32; m <<= 1) mx = fmaxf(mx, __shfl_xor(mx, m));
        float e = __expf(v - mx), s = e;
        for (int m = 1; m < 32; m <<= 1) s += __shfl_xor(s, m);
        logI[j] = __logf(e / s + 1e-10f);
    }
    for (int i = 0; i < 32; ++i) {
        float w = tl[i * 32 + j];
        float mx = w;
        for (int m = 1; m < 32; m <<= 1) mx = fmaxf(mx, __shfl_xor(mx, m));
        float e = __expf(w - mx), s = e;
        for (int m = 1; m < 32; m <<= 1) s += __shfl_xor(s, m);
        float p = e / s;
        tr[i * 32 + j]   = p;
        logT[i * 32 + j] = __logf(p + 1e-10f);
    }
}

// ---------------------------------------------------------------------------
// Fused emission MLP: x@W1+b1 -> LN -> GELU -> @W2+b2 -> GELU -> @W3+b3
// -> log_softmax.  2 waves/block, each wave owns a 16-token tile.
// All GEMMs via v_wmma_f32_16x16x32_f16.
// ---------------------------------------------------------------------------
__global__ __launch_bounds__(64) void emission_mlp(
    const float* __restrict__ x, const _Float16* __restrict__ wf,
    const float* __restrict__ b1, const float* __restrict__ lng,
    const float* __restrict__ lnb, const float* __restrict__ b2,
    const float* __restrict__ b3, float* __restrict__ em) {

    const int lane = threadIdx.x & 31;
    const int wv   = threadIdx.x >> 5;
    const int tok0 = (blockIdx.x * 2 + wv) * 16;
    const int m    = lane & 15;   // A-matrix row / lane%16
    const int g    = lane >> 4;   // half-wave group

    __shared__ __attribute__((aligned(16))) _Float16 stA[2][16][HID];
    __shared__ __attribute__((aligned(16))) _Float16 stB[2][16][HID];
    _Float16 (*s1)[HID] = stA[wv];
    _Float16 (*s2)[HID] = stB[wv];

    const _Float16* W1f = wf;                              // 4 kt x 16 nt tiles
    const _Float16* W2f = wf + INSZ * HID;                 // 8 kt x 16 nt
    const _Float16* W3f = wf + INSZ * HID + HID * HID;     // 8 kt x 2 nt

    // ---- stage 1: h1 = x @ W1 + b1, raw values to LDS, running stats ----
    v16h a1[4];
    const float* xr = x + (size_t)(tok0 + m) * INSZ;
    const v4f* xv = (const v4f*)xr;
    #pragma unroll
    for (int kt = 0; kt < 4; ++kt) {
        int k0 = kt * 32 + g * 8;
        v4f f0 = xv[k0 / 4], f1 = xv[k0 / 4 + 1];
        v4f f2 = xv[(k0 + 16) / 4], f3 = xv[(k0 + 16) / 4 + 1];
        v8h lo, hi;
        #pragma unroll
        for (int i = 0; i < 4; ++i) {
            lo[i] = (_Float16)f0[i]; lo[4 + i] = (_Float16)f1[i];
            hi[i] = (_Float16)f2[i]; hi[4 + i] = (_Float16)f3[i];
        }
        a1[kt] = joinh(lo, hi);
    }
    float rs[8], rq[8];
    #pragma unroll
    for (int r = 0; r < 8; ++r) { rs[r] = 0.f; rq[r] = 0.f; }

    for (int nt = 0; nt < 16; ++nt) {
        int n = nt * 16 + m;
        float bias = b1[n];
        v8f acc;
        #pragma unroll
        for (int r = 0; r < 8; ++r) acc[r] = bias;
        #pragma unroll
        for (int kt = 0; kt < 4; ++kt) {
            const _Float16* bptr = W1f + (size_t)((kt * 16 + nt) * 32 + lane) * 16;
            v16h bf = joinh(*(const v8h*)bptr, *(const v8h*)(bptr + 8));
            acc = __builtin_amdgcn_wmma_f32_16x16x32_f16(
                false, a1[kt], false, bf, (short)0, acc, false, false);
        }
        #pragma unroll
        for (int r = 0; r < 8; ++r) {
            float v = acc[r];
            rs[r] += v; rq[r] += v * v;
            s1[r + 8 * g][n] = (_Float16)v;
        }
    }
    // per-row mean/var (rows r+8g live on this lane's 16-lane group)
    float mu[8], rstd[8];
    #pragma unroll
    for (int r = 0; r < 8; ++r) {
        float a = rs[r], q = rq[r];
        for (int msk = 1; msk < 16; msk <<= 1) {
            a += __shfl_xor(a, msk);
            q += __shfl_xor(q, msk);
        }
        float mm = a * (1.0f / HID);
        float var = q * (1.0f / HID) - mm * mm;
        mu[r] = mm;
        rstd[r] = rsqrtf(var + 1e-5f);
    }
    // normalize + affine + GELU in place
    for (int nt = 0; nt < 16; ++nt) {
        int n = nt * 16 + m;
        float gg = lng[n], bb = lnb[n];
        #pragma unroll
        for (int r = 0; r < 8; ++r) {
            float v = (float)s1[r + 8 * g][n];
            v = (v - mu[r]) * rstd[r] * gg + bb;
            s1[r + 8 * g][n] = (_Float16)gelu_exact(v);
        }
    }

    // ---- stage 2: h2 = gelu(h1 @ W2 + b2) ----
    for (int nt = 0; nt < 16; ++nt) {
        int n = nt * 16 + m;
        float bias = b2[n];
        v8f acc;
        #pragma unroll
        for (int r = 0; r < 8; ++r) acc[r] = bias;
        #pragma unroll
        for (int kt = 0; kt < 8; ++kt) {
            int k0 = kt * 32 + g * 8;
            v16h af = joinh(*(const v8h*)&s1[m][k0], *(const v8h*)&s1[m][k0 + 16]);
            const _Float16* bptr = W2f + (size_t)((kt * 16 + nt) * 32 + lane) * 16;
            v16h bf = joinh(*(const v8h*)bptr, *(const v8h*)(bptr + 8));
            acc = __builtin_amdgcn_wmma_f32_16x16x32_f16(
                false, af, false, bf, (short)0, acc, false, false);
        }
        #pragma unroll
        for (int r = 0; r < 8; ++r)
            s2[r + 8 * g][n] = (_Float16)gelu_exact(acc[r]);
    }

    // ---- stage 3: logits = h2 @ W3 + b3, then log_softmax ----
    v8f acc3[2];
    #pragma unroll
    for (int nt = 0; nt < 2; ++nt) {
        float bias = b3[nt * 16 + m];
        #pragma unroll
        for (int r = 0; r < 8; ++r) acc3[nt][r] = bias;
        #pragma unroll
        for (int kt = 0; kt < 8; ++kt) {
            int k0 = kt * 32 + g * 8;
            v16h af = joinh(*(const v8h*)&s2[m][k0], *(const v8h*)&s2[m][k0 + 16]);
            const _Float16* bptr = W3f + (size_t)((kt * 2 + nt) * 32 + lane) * 16;
            v16h bf = joinh(*(const v8h*)bptr, *(const v8h*)(bptr + 8));
            acc3[nt] = __builtin_amdgcn_wmma_f32_16x16x32_f16(
                false, af, false, bf, (short)0, acc3[nt], false, false);
        }
    }
    #pragma unroll
    for (int r = 0; r < 8; ++r) {
        float mx = fmaxf(acc3[0][r], acc3[1][r]);
        for (int msk = 1; msk < 16; msk <<= 1) mx = fmaxf(mx, __shfl_xor(mx, msk));
        float se = __expf(acc3[0][r] - mx) + __expf(acc3[1][r] - mx);
        for (int msk = 1; msk < 16; msk <<= 1) se += __shfl_xor(se, msk);
        float lse = mx + __logf(se);
        size_t base = ((size_t)tok0 + r + 8 * g) * NSTATES;
        em[base + m]      = acc3[0][r] - lse;
        em[base + 16 + m] = acc3[1][r] - lse;
    }
}

// ---------------------------------------------------------------------------
// Forward + Viterbi scan. One wave per batch row, lane = state.
// Includes LDS-chunked backtrace.
// ---------------------------------------------------------------------------
__global__ __launch_bounds__(32) void hmm_scan(
    const float* __restrict__ em, const float* __restrict__ logT,
    const float* __restrict__ logI, float* __restrict__ sp,
    float* __restrict__ ll, float* __restrict__ ps,
    unsigned char* __restrict__ bp, float* __restrict__ bpath, int T) {

    const int b = blockIdx.x, j = threadIdx.x;
    __shared__ float sT[32 * 32];
    __shared__ float sA[32], sV[32];
    __shared__ unsigned int sBPd[64 * 8];

    for (int i = 0; i < 32; ++i) sT[i * 32 + j] = logT[i * 32 + j];
    __syncthreads();

    const size_t rb = (size_t)b * T;
    float alpha = logI[j] + em[rb * 32 + j];
    float vit = alpha;
    {
        float mx = alpha;
        for (int m = 1; m < 32; m <<= 1) mx = fmaxf(mx, __shfl_xor(mx, m));
        float e = __expf(alpha - mx), s = e;
        for (int m = 1; m < 32; m <<= 1) s += __shfl_xor(s, m);
        sp[rb * 32 + j] = e / s;
    }

    for (int t = 1; t < T; ++t) {
        float emv = em[(rb + t) * 32 + j];
        sA[j] = alpha; sV[j] = vit;
        __syncthreads();
        float mA = -3.4e38f, mV = -3.4e38f;
        int arg = 0;
        #pragma unroll
        for (int i = 0; i < 32; ++i) {
            float lt = sT[i * 32 + j];
            float a = sA[i] + lt;
            mA = fmaxf(mA, a);
            float v = sV[i] + lt;
            if (v > mV) { mV = v; arg = i; }   // first-index tie-break
        }
        float s = 0.f;
        #pragma unroll
        for (int i = 0; i < 32; ++i)
            s += __expf(sA[i] + sT[i * 32 + j] - mA);
        alpha = emv + mA + __logf(s);
        vit = emv + mV;
        bp[(rb + t) * 32 + j] = (unsigned char)arg;
        {
            float mx = alpha;
            for (int m = 1; m < 32; m <<= 1) mx = fmaxf(mx, __shfl_xor(mx, m));
            float e = __expf(alpha - mx), ss = e;
            for (int m = 1; m < 32; m <<= 1) ss += __shfl_xor(ss, m);
            sp[(rb + t) * 32 + j] = e / ss;
        }
        __syncthreads();
    }

    // log-likelihood
    {
        float mx = alpha;
        for (int m = 1; m < 32; m <<= 1) mx = fmaxf(mx, __shfl_xor(mx, m));
        float e = __expf(alpha - mx), s = e;
        for (int m = 1; m < 32; m <<= 1) s += __shfl_xor(s, m);
        if (j == 0) ll[b] = mx + __logf(s);
    }
    // Viterbi terminal: max + first-index argmax across lanes
    float pv = vit; int pi = j;
    for (int m = 1; m < 32; m <<= 1) {
        float ov = __shfl_xor(pv, m);
        int   oi = __shfl_xor(pi, m);
        if (ov > pv || (ov == pv && oi < pi)) { pv = ov; pi = oi; }
    }
    if (j == 0) ps[b] = pv;
    int state = pi;
    __threadfence();
    if (j == 0) bpath[rb + T - 1] = (float)state;

    // backtrace: chunk bp rows through LDS to avoid 8192 dependent HBM loads
    int t1 = T - 1;
    while (t1 >= 1) {
        int t0 = t1 - 63; if (t0 < 1) t0 = 1;
        int cnt = t1 - t0 + 1;
        const unsigned int* src = (const unsigned int*)(bp + (rb + t0) * 32);
        for (int q = j; q < cnt * 8; q += 32) sBPd[q] = src[q];
        __syncthreads();
        const unsigned char* sB = (const unsigned char*)sBPd;
        for (int t = t1; t >= t0; --t) {
            state = sB[(t - t0) * 32 + state];
            if (j == 0) bpath[rb + t - 1] = (float)state;
        }
        __syncthreads();
        t1 = t0 - 1;
    }
}

// ---------------------------------------------------------------------------
extern "C" void kernel_launch(void* const* d_in, const int* in_sizes, int n_in,
                              void* d_out, int out_size, void* d_ws, size_t ws_size,
                              hipStream_t stream) {
    (void)in_sizes; (void)n_in; (void)out_size; (void)ws_size;

    const float* x   = (const float*)d_in[0];
    const float* il  = (const float*)d_in[1];
    const float* tl  = (const float*)d_in[2];
    const float* W1  = (const float*)d_in[3];
    const float* b1  = (const float*)d_in[4];
    const float* lng = (const float*)d_in[5];
    const float* lnb = (const float*)d_in[6];
    const float* W2  = (const float*)d_in[7];
    const float* b2  = (const float*)d_in[8];
    const float* W3  = (const float*)d_in[9];
    const float* b3  = (const float*)d_in[10];

    const int B = BB, T = TT;
    const size_t TOK = (size_t)B * T;

    // output layout (tuple flattened)
    float* out   = (float*)d_out;
    float* em    = out;                       // (B,T,32)
    float* sp    = em + TOK * NSTATES;        // (B,T,32)
    float* ll    = sp + TOK * NSTATES;        // (B,)
    float* tr    = ll + B;                    // (32,32)
    float* bpath = tr + NSTATES * NSTATES;    // (B,T) as float
    float* ps    = bpath + TOK;               // (B,)

    // workspace layout
    char* ws = (char*)d_ws;
    _Float16* wfrag = (_Float16*)ws;                              // 212992 B
    float* logT = (float*)(ws + 212992);                          // 4096 B
    float* logI = (float*)(ws + 217088);                          // 128 B
    unsigned char* bp = (unsigned char*)(ws + 217216);            // B*T*32 B

    swizzleW<<<(INSZ * HID + 255) / 256, 256, 0, stream>>>(W1, wfrag, INSZ, HID);
    swizzleW<<<(HID * HID + 255) / 256, 256, 0, stream>>>(W2, wfrag + INSZ * HID, HID, HID);
    swizzleW<<<(HID * NSTATES + 255) / 256, 256, 0, stream>>>(
        W3, wfrag + INSZ * HID + HID * HID, HID, NSTATES);

    prep_hmm<<<1, 32, 0, stream>>>(il, tl, tr, logT, logI);

    // 2 waves/block, 16 tokens/wave -> 32 tokens/block
    emission_mlp<<<(int)(TOK / 32), 64, 0, stream>>>(x, wfrag, b1, lng, lnb, b2, b3, em);

    hmm_scan<<<B, 32, 0, stream>>>(em, logT, logI, sp, ll, ps, bp, bpath, T);
}
